// PointNet2_Encoder_v2_35553739277011
// MI455X (gfx1250) — matrix operand order repaired and checked by reading, not compiled, not run on hardware
//
#include <hip/hip_runtime.h>
#include <cstddef>

// ---------------- types ----------------
typedef __bf16 bf16;
typedef __bf16 v16bf __attribute__((ext_vector_type(16)));
typedef __bf16 v8bf  __attribute__((ext_vector_type(8)));
typedef float  v8f   __attribute__((ext_vector_type(8)));

#define BATCH 16
#define NPTS  8192
#define NP1   512
#define NP2   128

// ---------------- static device workspaces (compile-time shapes) ----------------
__device__ float g_pts [BATCH*NPTS*3];
__device__ float g_dist[BATCH*NPTS];
__device__ int   g_idx1[BATCH*NP1];
__device__ float g_nx1 [BATCH*NP1*3];
__device__ int   g_idx2[BATCH*NP2];
__device__ float g_nx2 [BATCH*NP2*3];
__device__ int   g_gidx[BATCH*NP1*128];
__device__ bf16  g_Xa[100663296];      // max M*Kp bf16 operand (1M x 96)
__device__ bf16  g_Xb[ 67108864];      // ping-pong buffer (1M x 64)
__device__ float g_Y [134217728];      // f32 GEMM output (1M x 128)
__device__ bf16  g_Wb[1024*672];       // bf16 weight staging
__device__ float g_part[256*1024*2];   // BN partial sums
__device__ float g_mean[1024];
__device__ float g_rstd[1024];
__device__ float g_feat1[BATCH*NP1*320];
__device__ float g_feat2[BATCH*NP2*640];
__device__ float g_feat3[BATCH*1024];
__device__ float g_headb[BATCH*1027];
__device__ float g_h1[BATCH*512];
__device__ float g_h2[BATCH*256];
__device__ float g_h3[BATCH*256];

// ---------------- kernels ----------------

// (B,3,N) -> (B,N,3)
__global__ void transpose_xyz_kernel(const float* __restrict__ in, float* __restrict__ out) {
  int i = blockIdx.x * blockDim.x + threadIdx.x;
  if (i >= BATCH * NPTS * 3) return;
  int c = i % 3; int n = (i / 3) % NPTS; int b = i / (3 * NPTS);
  out[i] = in[((size_t)b * 3 + c) * NPTS + n];
}

// Farthest point sampling: one workgroup per batch, sequential npoint iterations.
__global__ void fps_kernel(const float* __restrict__ pts, int Npts, int npoint,
                           float* __restrict__ dist, int* __restrict__ outIdx,
                           float* __restrict__ outXyz) {
  int b = blockIdx.x;
  const float* P = pts + (size_t)b * Npts * 3;
  float* D = dist + (size_t)b * Npts;
  __shared__ float scent[3];
  __shared__ float sval[1024];
  __shared__ int   sidx[1024];
  __shared__ int   sfar;
  int tid = threadIdx.x, nt = blockDim.x;
  for (int i = tid; i < Npts; i += nt) D[i] = 1e10f;
  if (tid == 0) sfar = 0;
  __syncthreads();
  for (int s = 0; s < npoint; ++s) {
    int far = sfar;
    if (tid == 0) {
      outIdx[b * npoint + s] = far;
      float cx = P[far*3+0], cy = P[far*3+1], cz = P[far*3+2];
      scent[0] = cx; scent[1] = cy; scent[2] = cz;
      outXyz[((size_t)b * npoint + s) * 3 + 0] = cx;
      outXyz[((size_t)b * npoint + s) * 3 + 1] = cy;
      outXyz[((size_t)b * npoint + s) * 3 + 2] = cz;
    }
    __syncthreads();
    float cx = scent[0], cy = scent[1], cz = scent[2];
    float best = -1.f; int bi = 0;
    for (int i = tid; i < Npts; i += nt) {
      float dx = P[i*3+0]-cx, dy = P[i*3+1]-cy, dz = P[i*3+2]-cz;
      float d = dx*dx + dy*dy + dz*dz;
      float dd = fminf(D[i], d);
      D[i] = dd;
      if (dd > best) { best = dd; bi = i; }   // strict > keeps first occurrence
    }
    sval[tid] = best; sidx[tid] = bi;
    __syncthreads();
    for (int off = nt >> 1; off > 0; off >>= 1) {
      if (tid < off) {
        float v2 = sval[tid+off]; int i2 = sidx[tid+off];
        if (v2 > sval[tid] || (v2 == sval[tid] && i2 < sidx[tid])) { sval[tid] = v2; sidx[tid] = i2; }
      }
      __syncthreads();
    }
    if (tid == 0) sfar = sidx[0];
    __syncthreads();
  }
}

// Ball query: one wave (32 lanes) per centroid; ordered compaction of first k indices.
__global__ __launch_bounds__(32)
void ballquery_kernel(const float* __restrict__ xyz, const float* __restrict__ cent,
                      int Npts, int npoint, int k, float r2, int* __restrict__ out) {
  int cidx = blockIdx.x;                 // b*npoint + s
  int b = cidx / npoint;
  const float* P = xyz + (size_t)b * Npts * 3;
  const float* C = cent + (size_t)cidx * 3;
  float cx = C[0], cy = C[1], cz = C[2];
  int lane = threadIdx.x;
  int* o = out + (size_t)cidx * k;
  int cnt = 0, first = 0;
  for (int base = 0; base < Npts && cnt < k; base += 32) {
    int n = base + lane;
    bool pred = false;
    if (n < Npts) {
      float dx = P[n*3+0]-cx, dy = P[n*3+1]-cy, dz = P[n*3+2]-cz;
      pred = (dx*dx + dy*dy + dz*dz) <= r2;
    }
    unsigned mask = (unsigned)__ballot(pred);
    if (cnt == 0 && mask) first = base + __ffs((int)mask) - 1;
    if (pred) {
      int pos = cnt + __popc(mask & ((1u << lane) - 1u));
      if (pos < k) o[pos] = n;
    }
    cnt += __popc(mask);
  }
  if (cnt > k) cnt = k;
  for (int p = cnt + lane; p < k; p += 32) o[p] = first;
}

// sa1 grouping: gxyz only, padded 3->32, bf16 output. One wave per row.
__global__ __launch_bounds__(32)
void group_xyz_kernel(const float* __restrict__ pts, const float* __restrict__ cent,
                      const int* __restrict__ gidx, int npoint, int k, int Npts,
                      bf16* __restrict__ X) {
  size_t row = blockIdx.x;               // ((b*npoint)+s)*k + j
  size_t cs = row / k;
  int b = (int)(cs / npoint);
  int gi = gidx[row];
  const float* p = pts + ((size_t)b * Npts + gi) * 3;
  const float* c = cent + cs * 3;
  int t = threadIdx.x;
  float v = (t < 3) ? (p[t] - c[t]) : 0.f;
  X[row * 32 + t] = (bf16)v;
}

// sa2 grouping: [feat(320) | gxyz(3) | pad->352]
__global__ void group_feat_kernel(const float* __restrict__ feat, int Cf,
                                  const float* __restrict__ xyz, const float* __restrict__ cent,
                                  const int* __restrict__ gidx, int npoint, int k, int Npts,
                                  bf16* __restrict__ X, int Kp) {
  size_t row = blockIdx.x;
  size_t cs = row / k;
  int b = (int)(cs / npoint);
  int gi = gidx[row];
  const float* f = feat + ((size_t)b * Npts + gi) * Cf;
  const float* p = xyz + ((size_t)b * Npts + gi) * 3;
  const float* c = cent + cs * 3;
  bf16* o = X + row * Kp;
  for (int t = threadIdx.x; t < Kp; t += blockDim.x) {
    float v = 0.f;
    if (t < Cf) v = f[t];
    else if (t < Cf + 3) v = p[t - Cf] - c[t - Cf];
    o[t] = (bf16)v;
  }
}

// sa3 group-all: [xyz(3) | feat(640) | pad->672]
__global__ void group_all_kernel(const float* __restrict__ xyz2, const float* __restrict__ feat2,
                                 bf16* __restrict__ X) {
  size_t row = blockIdx.x;               // b*128 + n
  bf16* o = X + row * 672;
  for (int t = threadIdx.x; t < 672; t += blockDim.x) {
    float v = 0.f;
    if (t < 3) v = xyz2[row * 3 + t];
    else if (t < 643) v = feat2[row * 640 + (t - 3)];
    o[t] = (bf16)v;
  }
}

// W f32 [N,K] -> bf16 [N,Kp] zero-padded
__global__ void w2bf_kernel(const float* __restrict__ W, bf16* __restrict__ o,
                            int N, int K, int Kp) {
  int i = blockIdx.x * blockDim.x + threadIdx.x;
  if (i >= N * Kp) return;
  int n = i / Kp, kk = i % Kp;
  o[i] = (bf16)(kk < K ? W[(size_t)n * K + kk] : 0.f);
}

// Y[M,N] = Xbf16[M,Kp] * Wbf16[N,Kp]^T + bias.
// One wave per 16 x (NT*16) output strip; NT compile-time so the K-loop is branch-free
// and every WMMA accumulates in-place (no accumulator register copies).
// Fragment layouts per CDNA5 ISA 7.12.2 (16-bit A 16x32, B 32x16, f32 C/D 16x16).
template <int NT>
__global__ __launch_bounds__(32)
void wmma_gemm_bias_kernel(const bf16* __restrict__ X, const bf16* __restrict__ W,
                           const float* __restrict__ bias, float* __restrict__ Y,
                           int M, int N, int Kp, int nbase) {
  const int lane = threadIdx.x;
  const int r    = lane & 15;
  const int hi   = lane >> 4;
  const int tileM = blockIdx.x * 16;
  const int tileN = nbase + blockIdx.y * (NT * 16);
  const bf16* xrow  = X + (size_t)(tileM + r) * Kp + 8 * hi;   // A: lane-group K offset
  const bf16* wbase = W + (size_t)(tileN + r) * Kp + 8 * hi;   // B: column r, lane-group K offset
  v8f acc[NT] = {};
  for (int k0 = 0; k0 < Kp; k0 += 32) {
    union { v16bf v; v8bf h[2]; } au;
    au.h[0] = *(const v8bf*)(xrow + k0);        // K = k0 + 8*hi    .. +8
    au.h[1] = *(const v8bf*)(xrow + k0 + 16);   // K = k0+16+8*hi  .. +8
    v16bf a = au.v;
#pragma unroll
    for (int t = 0; t < NT; ++t) {
      union { v16bf v; v8bf h[2]; } bu;
      bu.h[0] = *(const v8bf*)(wbase + (size_t)t * 16 * Kp + k0);
      bu.h[1] = *(const v8bf*)(wbase + (size_t)t * 16 * Kp + k0 + 16);
      v16bf bfrag = bu.v;
      acc[t] = __builtin_amdgcn_wmma_f32_16x16x32_bf16(
          false, a, false, bfrag, (short)0, acc[t], false, false);
    }
  }
#pragma unroll
  for (int t = 0; t < NT; ++t) {
    int col = tileN + t * 16 + r;
    float bv = bias[col];
#pragma unroll
    for (int i = 0; i < 8; ++i) {
      int m = tileM + i + 8 * hi;               // C/D layout: VGPR i -> M = i + 8*hi
      Y[(size_t)m * N + col] = acc[t][i] + bv;
    }
  }
}

// BN stats pass 1: 256 blocks, per-channel partial sum/sumsq
__global__ void colstats_part_kernel(const float* __restrict__ Y, int M, int C,
                                     float* __restrict__ part) {
  int blk = blockIdx.x;
  for (int c = threadIdx.x; c < C; c += blockDim.x) {
    float s = 0.f, q = 0.f;
    for (size_t r2 = blk; r2 < (size_t)M; r2 += gridDim.x) {
      float v = Y[r2 * C + c];
      s += v; q += v * v;
    }
    part[((size_t)blk * C + c) * 2 + 0] = s;
    part[((size_t)blk * C + c) * 2 + 1] = q;
  }
}

// BN stats pass 2: one block per channel
__global__ void colstats_final_kernel(const float* __restrict__ part, int M, int C,
                                      float* __restrict__ mean, float* __restrict__ rstd) {
  int c = blockIdx.x;
  __shared__ float ss[256], sq[256];
  float s = 0.f, q = 0.f;
  for (int i = threadIdx.x; i < 256; i += blockDim.x) {
    s += part[((size_t)i * C + c) * 2 + 0];
    q += part[((size_t)i * C + c) * 2 + 1];
  }
  ss[threadIdx.x] = s; sq[threadIdx.x] = q;
  __syncthreads();
  for (int off = blockDim.x >> 1; off > 0; off >>= 1) {
    if (threadIdx.x < off) { ss[threadIdx.x] += ss[threadIdx.x+off]; sq[threadIdx.x] += sq[threadIdx.x+off]; }
    __syncthreads();
  }
  if (threadIdx.x == 0) {
    float m = ss[0] / (float)M;
    float v = sq[0] / (float)M - m * m;
    mean[c] = m;
    rstd[c] = rsqrtf(v + 1e-5f);
  }
}

// normalize + affine + relu; in-place f32 and optional bf16 for next GEMM
__global__ void bnrelu_kernel(float* __restrict__ Y, const float* __restrict__ mean,
                              const float* __restrict__ rstd, const float* __restrict__ g,
                              const float* __restrict__ be, size_t total, int C,
                              bf16* __restrict__ Xout) {
  size_t i = (size_t)blockIdx.x * blockDim.x + threadIdx.x;
  if (i >= total) return;
  int c = (int)(i % C);
  float x = (Y[i] - mean[c]) * rstd[c] * g[c] + be[c];
  x = fmaxf(x, 0.f);
  Y[i] = x;
  if (Xout) Xout[i] = (bf16)x;
}

// max over the k grouped samples; writes into channel-offset slice of the concat output
__global__ void maxpool_kernel(const float* __restrict__ Y, int k, int C,
                               float* __restrict__ out, int Ctot, int coff) {
  int cs = blockIdx.x;                   // b*npoint + s
  const float* y = Y + (size_t)cs * k * C;
  for (int c = threadIdx.x; c < C; c += blockDim.x) {
    float m = -3.4e38f;
    for (int j = 0; j < k; ++j) m = fmaxf(m, y[(size_t)j * C + c]);
    out[(size_t)cs * Ctot + coff + c] = m;
  }
}

__global__ void build_head_kernel(const float* __restrict__ feat3, float* __restrict__ out) {
  int i = blockIdx.x * blockDim.x + threadIdx.x;
  if (i >= BATCH * 1027) return;
  int b = i / 1027, c = i % 1027;
  out[i] = (c < 3) ? 0.f : feat3[b * 1024 + (c - 3)];
}

// tiny f32 GEMM for the head: Y[b,n] = X[b,:]·W[n,:] + bias[n]
__global__ void head_gemm_kernel(const float* __restrict__ X, const float* __restrict__ W,
                                 const float* __restrict__ bvec, float* __restrict__ Y,
                                 int K, int N) {
  int b = blockIdx.y;
  int n = blockIdx.x * blockDim.x + threadIdx.x;
  if (n >= N) return;
  const float* x = X + (size_t)b * K;
  const float* w = W + (size_t)n * K;
  float s = bvec[n];
  for (int i = 0; i < K; ++i) s += x[i] * w[i];
  Y[(size_t)b * N + n] = s;
}

__global__ void instnorm_kernel(const float* __restrict__ in, float* __restrict__ out,
                                int N, int leaky) {
  int b = blockIdx.x;
  __shared__ float ss[256], sq[256];
  float s = 0.f, q = 0.f;
  for (int i = threadIdx.x; i < N; i += blockDim.x) {
    float v = in[(size_t)b * N + i]; s += v; q += v * v;
  }
  ss[threadIdx.x] = s; sq[threadIdx.x] = q;
  __syncthreads();
  for (int off = blockDim.x >> 1; off > 0; off >>= 1) {
    if (threadIdx.x < off) { ss[threadIdx.x] += ss[threadIdx.x+off]; sq[threadIdx.x] += sq[threadIdx.x+off]; }
    __syncthreads();
  }
  float m = ss[0] / (float)N;
  float var = sq[0] / (float)N - m * m;
  float rs = rsqrtf(var + 1e-5f);
  for (int i = threadIdx.x; i < N; i += blockDim.x) {
    float x = (in[(size_t)b * N + i] - m) * rs;
    if (leaky) x = (x > 0.f) ? x : 0.01f * x;
    out[(size_t)b * N + i] = x;
  }
}

// ---------------- host side ----------------

struct BNL { const float *W, *b, *g, *be; int cout; };

static void launch_gemm(const bf16* X, const bf16* W, const float* b, float* Y,
                        int M, int N, int Kp, hipStream_t st) {
  int full = N / 64;
  if (full > 0)
    wmma_gemm_bias_kernel<4><<<dim3((unsigned)(M / 16), (unsigned)full), 32, 0, st>>>(
        X, W, b, Y, M, N, Kp, 0);
  int rem = (N % 64) / 16;
  int nbase = full * 64;
  dim3 tg((unsigned)(M / 16), 1);
  if (rem == 3)      wmma_gemm_bias_kernel<3><<<tg, 32, 0, st>>>(X, W, b, Y, M, N, Kp, nbase);
  else if (rem == 2) wmma_gemm_bias_kernel<2><<<tg, 32, 0, st>>>(X, W, b, Y, M, N, Kp, nbase);
  else if (rem == 1) wmma_gemm_bias_kernel<1><<<tg, 32, 0, st>>>(X, W, b, Y, M, N, Kp, nbase);
}

static void run_chain(bf16* Xin, bf16* Xalt, int M, int Kp, int Kreal,
                      const BNL* L, int nl, float* Y, bf16* Wb,
                      float* part, float* mean, float* rstd, hipStream_t st) {
  for (int l = 0; l < nl; ++l) {
    int C = L[l].cout;
    int tot = C * Kp;
    w2bf_kernel<<<(tot + 255) / 256, 256, 0, st>>>(L[l].W, Wb, C, Kreal, Kp);
    launch_gemm(Xin, Wb, L[l].b, Y, M, C, Kp, st);
    colstats_part_kernel<<<256, 256, 0, st>>>(Y, M, C, part);
    colstats_final_kernel<<<C, 256, 0, st>>>(part, M, C, mean, rstd);
    size_t total = (size_t)M * C;
    bf16* Xout = (l == nl - 1) ? (bf16*)nullptr : Xalt;
    bnrelu_kernel<<<(unsigned)((total + 255) / 256), 256, 0, st>>>(
        Y, mean, rstd, L[l].g, L[l].be, total, C, Xout);
    if (l != nl - 1) { bf16* t = Xin; Xin = Xalt; Xalt = t; Kp = C; Kreal = C; }
  }
}

extern "C" void kernel_launch(void* const* d_in, const int* in_sizes, int n_in,
                              void* d_out, int out_size, void* d_ws, size_t ws_size,
                              hipStream_t stream) {
  (void)d_ws; (void)ws_size; (void)out_size;

  // ---- resolve static workspace symbols ----
  float *pts, *dist, *nx1, *nx2, *Y, *part, *mean, *rstd;
  float *feat1, *feat2, *feat3, *headb, *h1, *h2, *h3;
  int *idx1, *idx2, *gidx;
  bf16 *Xa, *Xb, *Wb;
  hipGetSymbolAddress((void**)&pts,   HIP_SYMBOL(g_pts));
  hipGetSymbolAddress((void**)&dist,  HIP_SYMBOL(g_dist));
  hipGetSymbolAddress((void**)&idx1,  HIP_SYMBOL(g_idx1));
  hipGetSymbolAddress((void**)&nx1,   HIP_SYMBOL(g_nx1));
  hipGetSymbolAddress((void**)&idx2,  HIP_SYMBOL(g_idx2));
  hipGetSymbolAddress((void**)&nx2,   HIP_SYMBOL(g_nx2));
  hipGetSymbolAddress((void**)&gidx,  HIP_SYMBOL(g_gidx));
  hipGetSymbolAddress((void**)&Xa,    HIP_SYMBOL(g_Xa));
  hipGetSymbolAddress((void**)&Xb,    HIP_SYMBOL(g_Xb));
  hipGetSymbolAddress((void**)&Y,     HIP_SYMBOL(g_Y));
  hipGetSymbolAddress((void**)&Wb,    HIP_SYMBOL(g_Wb));
  hipGetSymbolAddress((void**)&part,  HIP_SYMBOL(g_part));
  hipGetSymbolAddress((void**)&mean,  HIP_SYMBOL(g_mean));
  hipGetSymbolAddress((void**)&rstd,  HIP_SYMBOL(g_rstd));
  hipGetSymbolAddress((void**)&feat1, HIP_SYMBOL(g_feat1));
  hipGetSymbolAddress((void**)&feat2, HIP_SYMBOL(g_feat2));
  hipGetSymbolAddress((void**)&feat3, HIP_SYMBOL(g_feat3));
  hipGetSymbolAddress((void**)&headb, HIP_SYMBOL(g_headb));
  hipGetSymbolAddress((void**)&h1,    HIP_SYMBOL(g_h1));
  hipGetSymbolAddress((void**)&h2,    HIP_SYMBOL(g_h2));
  hipGetSymbolAddress((void**)&h3,    HIP_SYMBOL(g_h3));

  // ---- unpack inputs (xyz + pytree of params: head, sa1, sa2, sa3) ----
  const float* xyz_in;
  int pi;
  if (n_in > 0 && in_sizes[0] == BATCH * 3 * NPTS) { xyz_in = (const float*)d_in[0]; pi = 1; }
  else { xyz_in = (const float*)d_in[n_in - 1]; pi = 0; }
  auto nxt = [&]() { return (const float*)d_in[pi++]; };

  const float *hW[3], *hb[3];
  for (int i = 0; i < 3; ++i) { hW[i] = nxt(); hb[i] = nxt(); }

  static const int ch1[3][3] = {{32,32,64},{64,64,128},{64,96,128}};
  static const int ch2[3][3] = {{64,64,128},{128,128,256},{128,128,256}};
  static const int ch3[3]    = {256,512,1024};
  BNL sa1L[3][3], sa2L[3][3], sa3L[3];
  for (int s = 0; s < 3; ++s)
    for (int l = 0; l < 3; ++l) {
      sa1L[s][l].W = nxt(); sa1L[s][l].b = nxt(); sa1L[s][l].g = nxt(); sa1L[s][l].be = nxt();
      sa1L[s][l].cout = ch1[s][l];
    }
  for (int s = 0; s < 3; ++s)
    for (int l = 0; l < 3; ++l) {
      sa2L[s][l].W = nxt(); sa2L[s][l].b = nxt(); sa2L[s][l].g = nxt(); sa2L[s][l].be = nxt();
      sa2L[s][l].cout = ch2[s][l];
    }
  for (int l = 0; l < 3; ++l) {
    sa3L[l].W = nxt(); sa3L[l].b = nxt(); sa3L[l].g = nxt(); sa3L[l].be = nxt();
    sa3L[l].cout = ch3[l];
  }

  // ---- stage 0: transpose + FPS level 1 ----
  transpose_xyz_kernel<<<(BATCH*NPTS*3 + 255) / 256, 256, 0, stream>>>(xyz_in, pts);
  fps_kernel<<<BATCH, 1024, 0, stream>>>(pts, NPTS, NP1, dist, idx1, nx1);

  // ---- sa1 (MSG) ----
  static const int   k1[3]   = {16, 32, 128};
  static const float r1sq[3] = {0.01f, 0.04f, 0.16f};
  int coff = 0;
  for (int s = 0; s < 3; ++s) {
    int k = k1[s];
    int M = BATCH * NP1 * k;
    ballquery_kernel<<<BATCH * NP1, 32, 0, stream>>>(pts, nx1, NPTS, NP1, k, r1sq[s], gidx);
    group_xyz_kernel<<<M, 32, 0, stream>>>(pts, nx1, gidx, NP1, k, NPTS, Xa);
    run_chain(Xa, Xb, M, 32, 3, sa1L[s], 3, Y, Wb, part, mean, rstd, stream);
    int C = ch1[s][2];
    maxpool_kernel<<<BATCH * NP1, 128, 0, stream>>>(Y, k, C, feat1, 320, coff);
    coff += C;
  }

  // ---- sa2 (MSG on the 512 FPS points) ----
  fps_kernel<<<BATCH, 512, 0, stream>>>(nx1, NP1, NP2, dist, idx2, nx2);
  static const int   k2[3]   = {32, 64, 128};
  static const float r2sq[3] = {0.04f, 0.16f, 0.64f};
  coff = 0;
  for (int s = 0; s < 3; ++s) {
    int k = k2[s];
    int M = BATCH * NP2 * k;
    ballquery_kernel<<<BATCH * NP2, 32, 0, stream>>>(nx1, nx2, NP1, NP2, k, r2sq[s], gidx);
    group_feat_kernel<<<M, 64, 0, stream>>>(feat1, 320, nx1, nx2, gidx, NP2, k, NP1, Xa, 352);
    run_chain(Xa, Xb, M, 352, 323, sa2L[s], 3, Y, Wb, part, mean, rstd, stream);
    int C = ch2[s][2];
    maxpool_kernel<<<BATCH * NP2, 128, 0, stream>>>(Y, k, C, feat2, 640, coff);
    coff += C;
  }

  // ---- sa3 (group-all) ----
  group_all_kernel<<<BATCH * NP2, 64, 0, stream>>>(nx2, feat2, Xa);
  run_chain(Xa, Xb, BATCH * NP2, 672, 643, sa3L, 3, Y, Wb, part, mean, rstd, stream);
  maxpool_kernel<<<BATCH, 256, 0, stream>>>(Y, NP2, 1024, feat3, 1024, 0);

  // ---- head: [zeros(3)|feat(1024)] -> 512 -> 256 -> 256 with inst_norm, leaky at end ----
  build_head_kernel<<<(BATCH * 1027 + 255) / 256, 256, 0, stream>>>(feat3, headb);
  head_gemm_kernel<<<dim3(2, BATCH), 256, 0, stream>>>(headb, hW[0], hb[0], h1, 1027, 512);
  instnorm_kernel<<<BATCH, 256, 0, stream>>>(h1, h1, 512, 0);
  head_gemm_kernel<<<dim3(1, BATCH), 256, 0, stream>>>(h1, hW[1], hb[1], h2, 512, 256);
  instnorm_kernel<<<BATCH, 256, 0, stream>>>(h2, h2, 256, 0);
  head_gemm_kernel<<<dim3(1, BATCH), 256, 0, stream>>>(h2, hW[2], hb[2], h3, 256, 256);
  instnorm_kernel<<<BATCH, 256, 0, stream>>>(h3, (float*)d_out, 256, 1);
}
